// EncoderLayer_53772990546005
// MI455X (gfx1250) — compile-verified, matrix operand order checked by reading, not run on hardware
//
#include <hip/hip_runtime.h>

// ---------------------------------------------------------------------------
// Problem constants (from reference)
// ---------------------------------------------------------------------------
#define BATCH   256
#define NTOK    181
#define DMODEL  512
#define HEADS   8
#define DKH     64
#define DFF     2048
#define NVV     121
#define NCC     60          // NTOK - NVV
#define MROWS   (BATCH * NTOK)   // 46336, divisible by 128
#define NPADV   192         // padded key-column dim of transposed V
#define VCOLCTX 128         // context tokens remapped to col 128.. (16B aligned)

typedef __bf16 bf16_t;
typedef __attribute__((ext_vector_type(16))) __bf16 v16bf;
typedef __attribute__((ext_vector_type(8)))  __bf16 v8bf;
typedef __attribute__((ext_vector_type(8)))  float  v8f;

#define WMMA_BF16(a, b, c)                                                    \
  __builtin_amdgcn_wmma_f32_16x16x32_bf16(false, (a), false, (b), (short)0,   \
                                          (c), false, false)

// float -> bf16 (round to nearest even)
__device__ __forceinline__ bf16_t f2b(float f) {
  unsigned x = __builtin_bit_cast(unsigned, f);
  unsigned r = (x + 0x7FFFu + ((x >> 16) & 1u)) >> 16;
  unsigned short s = (unsigned short)r;
  return __builtin_bit_cast(bf16_t, s);
}

// Load one 16x32 bf16 WMMA operand fragment (A-style swizzle, ISA 7.12.2):
//   lanes 0..15  (sub=0): row = lane,     K chunks {0..7} and {16..23}
//   lanes 16..31 (sub=1): row = lane-16,  K chunks {8..15} and {24..31}
// Works for global or LDS pointers (row offsets must be 8-element aligned).
__device__ __forceinline__ v16bf load_frag(const bf16_t* base, int row_stride,
                                           int row, int kbase, int sub) {
  const bf16_t* p = base + (size_t)row * row_stride + kbase + sub * 8;
  union { v16bf v; v8bf h[2]; } u;
  u.h[0] = *(const v8bf*)(p);
  u.h[1] = *(const v8bf*)(p + 16);
  return u.v;
}

// ---------------------------------------------------------------------------
// fp32 -> bf16 conversion
// ---------------------------------------------------------------------------
__global__ void cvt_kernel(const float* __restrict__ in,
                           bf16_t* __restrict__ out, int n) {
  int i = blockIdx.x * blockDim.x + threadIdx.x;
  if (i < n) out[i] = f2b(i < n ? in[i] : 0.0f);
}

// ---------------------------------------------------------------------------
// WMMA GEMM:  C[M,Nout] = A[M,K](bf16) * W[Nout,K](bf16)^T + bias
// Workgroup: 256 threads = 8 waves, tile 128(M) x 256(N); each wave 64x64
// (4x4 fragments of 16x16 => 16 v_wmma per 16 b128 loads per K step).
// MODE 0: row-major out (WF -> f32, WB -> bf16), RELU optional
// MODE 1: bf16 out in [B,H,N,DK] head layout (Q / K)
// MODE 2: bf16 out in [B,H,DK,NPADV] transposed-V layout (context cols at 128)
// MODE/RELU/WF/WB are compile-time so each instantiation keeps exactly one
// epilogue path (avoids accumulator spills seen with a runtime-mode epilogue).
// ---------------------------------------------------------------------------
template <int MODE, bool RELU, bool WF, bool WB>
__global__ __launch_bounds__(256) void gemm_bf16_kernel(
    const bf16_t* __restrict__ A, const bf16_t* __restrict__ W,
    const float* __restrict__ bias, float* __restrict__ outF,
    bf16_t* __restrict__ outB, int Nout, int Kdim) {
  const int tid  = threadIdx.x;
  const int lane = tid & 31;
  const int wid  = tid >> 5;  // 0..7
  const int wm   = wid & 1;   // 2 M sub-tiles of 64
  const int wn   = wid >> 1;  // 4 N sub-tiles of 64
  const int m0   = blockIdx.x * 128 + wm * 64;
  const int n0   = blockIdx.y * 256 + wn * 64;
  const int sub  = lane >> 4;
  const int rrow = lane & 15;

  v8f acc[4][4] = {};

  const bf16_t* arow = A + (size_t)(m0 + rrow) * Kdim + sub * 8;
  const bf16_t* wrow = W + (size_t)(n0 + rrow) * Kdim + sub * 8;

  for (int kb = 0; kb < Kdim; kb += 32) {
    // Speculative prefetch of the next K slice (dropped past end of buffer).
    __builtin_prefetch(arow + kb + 32, 0, 3);
    __builtin_prefetch(wrow + kb + 32, 0, 3);
    v16bf af[4], bf[4];
#pragma unroll
    for (int i = 0; i < 4; ++i)
      af[i] = load_frag(A, Kdim, m0 + i * 16 + rrow, kb, sub);
#pragma unroll
    for (int j = 0; j < 4; ++j)
      bf[j] = load_frag(W, Kdim, n0 + j * 16 + rrow, kb, sub);
#pragma unroll
    for (int i = 0; i < 4; ++i)
#pragma unroll
      for (int j = 0; j < 4; ++j) acc[i][j] = WMMA_BF16(af[i], bf[j], acc[i][j]);
  }

  // Epilogue. C/D layout: n = lane&15 ; m = r + 8*(lane>>4)
#pragma unroll
  for (int i = 0; i < 4; ++i) {
#pragma unroll
    for (int j = 0; j < 4; ++j) {
      const int col = n0 + j * 16 + rrow;
      const float bv = bias[col];
#pragma unroll
      for (int r = 0; r < 8; ++r) {
        float v = acc[i][j][r] + bv;
        if (RELU) v = fmaxf(v, 0.0f);
        const int m = m0 + i * 16 + r + sub * 8;
        if (MODE == 0) {
          const size_t off = (size_t)m * Nout + col;
          if (WF) outF[off] = v;
          if (WB) outB[off] = f2b(v);
        } else {
          const int bb = m / NTOK, nt = m % NTOK;
          const int hh = col >> 6, dd = col & 63;
          if (MODE == 1) {
            outB[(((size_t)bb * HEADS + hh) * NTOK + nt) * DKH + dd] = f2b(v);
          } else {
            const int nc = (nt < NVV) ? nt : (VCOLCTX + (nt - NVV));
            outB[(((size_t)bb * HEADS + hh) * DKH + dd) * NPADV + nc] = f2b(v);
          }
        }
      }
    }
  }
}

// ---------------------------------------------------------------------------
// Attention: one wave per (batch, head, 16-query tile).
//   scores = Q K^T / 8  (WMMA, K-dim 64) -> LDS, mask (+ -inf key padding),
//   per-row softmax, bf16 P in LDS, ctx = P V^T (WMMA, A-frags from LDS).
// ---------------------------------------------------------------------------
__global__ __launch_bounds__(32) void attn_kernel(
    const bf16_t* __restrict__ Q, const bf16_t* __restrict__ Kb,
    const bf16_t* __restrict__ Vt, const int* __restrict__ mask,
    bf16_t* __restrict__ ctx, int q_start, int q_len, int k_start, int k_len,
    int v_col_start, int h_start, int kpad, int kshift, int tmask) {
  __shared__ __attribute__((aligned(16))) float  sc[16 * 128];
  __shared__ __attribute__((aligned(16))) bf16_t pb[16 * 128];

  const int lane = threadIdx.x;
  const int sub  = lane >> 4;
  const int rrow = lane & 15;
  const int b    = blockIdx.z;
  const int hh   = h_start + blockIdx.y;
  const int m0   = blockIdx.x * 16;

  const bf16_t* Qh = Q  + (((size_t)b * HEADS + hh) * NTOK) * DKH;
  const bf16_t* Kh = Kb + (((size_t)b * HEADS + hh) * NTOK) * DKH;
  const bf16_t* Vh = Vt + (((size_t)b * HEADS + hh) * DKH) * NPADV;

  // ---- scores -> LDS ----
  const int kt = kpad >> 4;
  const int qrow = q_start + min(m0 + rrow, q_len - 1);
  for (int t = 0; t < kt; ++t) {
    v8f acc = {};
    const int krow = k_start + min(t * 16 + rrow, k_len - 1);
#pragma unroll
    for (int ks = 0; ks < DKH; ks += 32) {
      v16bf aq = load_frag(Qh, DKH, qrow, ks, sub);
      v16bf bk = load_frag(Kh, DKH, krow, ks, sub);
      acc = WMMA_BF16(aq, bk, acc);
    }
#pragma unroll
    for (int r = 0; r < 8; ++r) {
      const int row = r + sub * 8;
      sc[row * kpad + t * 16 + rrow] = acc[r] * 0.125f;  // 1/sqrt(64)
    }
  }
  __syncthreads();

  // ---- mask + key padding ----
  for (int idx = lane; idx < 16 * kpad; idx += 32) {
    const int qi = idx >> kshift;
    const int ki = idx & (kpad - 1);
    if (ki >= k_len) { sc[idx] = -__builtin_inff(); continue; }
    const int qt = q_start + m0 + qi;
    if (m0 + qi >= q_len || qt >= NTOK) continue;
    const int kk = k_start + ki;
    const int mv = tmask ? mask[kk * NTOK + qt] : mask[qt * NTOK + kk];
    if (mv) sc[idx] = -1.0e9f;
  }
  __syncthreads();

  // ---- softmax (one row per lane<16) ----
  if (lane < 16) {
    float mx = -__builtin_inff();
    for (int k = 0; k < kpad; ++k) mx = fmaxf(mx, sc[lane * kpad + k]);
    float sum = 0.0f;
    for (int k = 0; k < kpad; ++k) {
      float e = __expf(sc[lane * kpad + k] - mx);
      sc[lane * kpad + k] = e;
      sum += e;
    }
    const float inv = 1.0f / sum;
    for (int k = 0; k < kpad; ++k) pb[lane * kpad + k] = f2b(sc[lane * kpad + k] * inv);
  }
  __syncthreads();

  // ---- ctx = P * V^T ----
#pragma unroll
  for (int dt = 0; dt < DKH / 16; ++dt) {
    v8f acc = {};
    for (int kb2 = 0; kb2 < kpad; kb2 += 32) {
      v16bf ap = load_frag(&pb[0], kpad, rrow, kb2, sub);           // ds_load
      v16bf bv = load_frag(Vh, NPADV, dt * 16 + rrow, v_col_start + kb2, sub);
      acc = WMMA_BF16(ap, bv, acc);
    }
#pragma unroll
    for (int r = 0; r < 8; ++r) {
      const int qi = r + sub * 8;
      if (m0 + qi < q_len) {
        const int tok = q_start + m0 + qi;
        size_t off = ((size_t)b * NTOK + tok) * DMODEL + hh * DKH + dt * 16 + rrow;
        ctx[off] = f2b(acc[r]);
      }
    }
  }
}

// ---------------------------------------------------------------------------
// out = LayerNorm(X + Y) * g + b ; optional bf16 mirror for next GEMM
// ---------------------------------------------------------------------------
__global__ __launch_bounds__(256) void residual_ln_kernel(
    const float* __restrict__ X, const float* __restrict__ Y,
    const float* __restrict__ g, const float* __restrict__ be,
    float* __restrict__ outF, bf16_t* __restrict__ outB) {
  __shared__ float s1[256], s2[256];
  const int row = blockIdx.x;
  const int t = threadIdx.x;
  const float* xr = X + (size_t)row * DMODEL;
  const float* yr = Y + (size_t)row * DMODEL;
  const float v0 = xr[t] + yr[t];
  const float v1 = xr[t + 256] + yr[t + 256];
  s1[t] = v0 + v1;
  s2[t] = v0 * v0 + v1 * v1;
  __syncthreads();
  for (int s = 128; s > 0; s >>= 1) {
    if (t < s) { s1[t] += s1[t + s]; s2[t] += s2[t + s]; }
    __syncthreads();
  }
  const float mu  = s1[0] * (1.0f / DMODEL);
  const float var = s2[0] * (1.0f / DMODEL) - mu * mu;
  const float inv = rsqrtf(var + 1e-5f);
  const float o0 = (v0 - mu) * inv * g[t] + be[t];
  const float o1 = (v1 - mu) * inv * g[t + 256] + be[t + 256];
  float* orow = outF + (size_t)row * DMODEL;
  orow[t] = o0;
  orow[t + 256] = o1;
  if (outB) {
    bf16_t* ob = outB + (size_t)row * DMODEL;
    ob[t] = f2b(o0);
    ob[t + 256] = f2b(o1);
  }
}

// ---------------------------------------------------------------------------
// Workspace layout (bytes), with reuse. Peak ~478 MB.
// ---------------------------------------------------------------------------
#define OFF_WQ  ((size_t)0)
#define OFF_WK  ((size_t)524288)
#define OFF_WV  ((size_t)1048576)
#define OFF_WO  ((size_t)1572864)
#define OFF_W1  ((size_t)2097152)    // 2 MB
#define OFF_W2  ((size_t)4194304)    // 2 MB
#define OFF_XB  ((size_t)8  << 20)   // 45.3 MB bf16 x
#define OFF_Q   ((size_t)56 << 20)   // 45.3 MB
#define OFF_K   ((size_t)104 << 20)  // 45.3 MB
#define OFF_VT  ((size_t)152 << 20)  // 48.0 MB
#define OFF_CTX ((size_t)200 << 20)  // 45.3 MB
#define OFF_AO  ((size_t)248 << 20)  // 90.5 MB f32
#define OFF_H   ((size_t)340 << 20)  // 90.5 MB f32
#define OFF_HB  ((size_t)432 << 20)  // 45.3 MB
#define OFF_F1  OFF_Q                // 181 MB, reuses Q/K/Vt/CTX (dead by then)
#define OFF_FF  OFF_AO               // 90.5 MB, reuses attn_out (dead)

extern "C" void kernel_launch(void* const* d_in, const int* in_sizes, int n_in,
                              void* d_out, int out_size, void* d_ws,
                              size_t ws_size, hipStream_t stream) {
  (void)in_sizes; (void)n_in; (void)out_size; (void)ws_size;
  const float* x    = (const float*)d_in[0];
  const int*   mask = (const int*)d_in[1];
  const float* Wq = (const float*)d_in[2];  const float* bq = (const float*)d_in[3];
  const float* Wk = (const float*)d_in[4];  const float* bk = (const float*)d_in[5];
  const float* Wv = (const float*)d_in[6];  const float* bv = (const float*)d_in[7];
  const float* Wo = (const float*)d_in[8];  const float* bo = (const float*)d_in[9];
  const float* w1 = (const float*)d_in[10]; const float* b1 = (const float*)d_in[11];
  const float* w2 = (const float*)d_in[12]; const float* b2 = (const float*)d_in[13];
  const float* g1 = (const float*)d_in[14]; const float* be1 = (const float*)d_in[15];
  const float* g2 = (const float*)d_in[16]; const float* be2 = (const float*)d_in[17];

  char* ws = (char*)d_ws;
  bf16_t* wqb = (bf16_t*)(ws + OFF_WQ);
  bf16_t* wkb = (bf16_t*)(ws + OFF_WK);
  bf16_t* wvb = (bf16_t*)(ws + OFF_WV);
  bf16_t* wob = (bf16_t*)(ws + OFF_WO);
  bf16_t* w1b = (bf16_t*)(ws + OFF_W1);
  bf16_t* w2b = (bf16_t*)(ws + OFF_W2);
  bf16_t* xb  = (bf16_t*)(ws + OFF_XB);
  bf16_t* Qb  = (bf16_t*)(ws + OFF_Q);
  bf16_t* Kbf = (bf16_t*)(ws + OFF_K);
  bf16_t* Vtb = (bf16_t*)(ws + OFF_VT);
  bf16_t* ctx = (bf16_t*)(ws + OFF_CTX);
  float*  ao  = (float*)(ws + OFF_AO);
  float*  hF  = (float*)(ws + OFF_H);
  bf16_t* hB  = (bf16_t*)(ws + OFF_HB);
  bf16_t* f1b = (bf16_t*)(ws + OFF_F1);
  float*  ffF = (float*)(ws + OFF_FF);

  // ---- fp32 -> bf16 ----
  auto cv = [&](const float* src, bf16_t* dst, int n) {
    cvt_kernel<<<(n + 255) / 256, 256, 0, stream>>>(src, dst, n);
  };
  cv(x, xb, MROWS * DMODEL);
  cv(Wq, wqb, DMODEL * DMODEL);  cv(Wk, wkb, DMODEL * DMODEL);
  cv(Wv, wvb, DMODEL * DMODEL);  cv(Wo, wob, DMODEL * DMODEL);
  cv(w1, w1b, DFF * DMODEL);     cv(w2, w2b, DMODEL * DFF);

  const dim3 blk(256);
  const dim3 gD(MROWS / 128, DMODEL / 256);   // Nout = 512
  const dim3 gF(MROWS / 128, DFF / 256);      // Nout = 2048

  // ---- Q, K, V projections ----
  gemm_bf16_kernel<1, false, false, true><<<gD, blk, 0, stream>>>(
      xb, wqb, bq, nullptr, Qb, DMODEL, DMODEL);
  gemm_bf16_kernel<1, false, false, true><<<gD, blk, 0, stream>>>(
      xb, wkb, bk, nullptr, Kbf, DMODEL, DMODEL);
  gemm_bf16_kernel<2, false, false, true><<<gD, blk, 0, stream>>>(
      xb, wvb, bv, nullptr, Vtb, DMODEL, DMODEL);

  // ---- attention (4 segment cases) ----
  // r1v: heads 0-3, q=visual, kv=context
  attn_kernel<<<dim3(8, 4, BATCH), 32, 0, stream>>>(
      Qb, Kbf, Vtb, mask, ctx, 0, NVV, NVV, NCC, VCOLCTX, 0, 64, 6, 0);
  // r1c: heads 0-3, q=context, kv=visual (transposed mask)
  attn_kernel<<<dim3(4, 4, BATCH), 32, 0, stream>>>(
      Qb, Kbf, Vtb, mask, ctx, NVV, NCC, 0, NVV, 0, 0, 128, 7, 1);
  // r2v: heads 4-7, visual self-attn
  attn_kernel<<<dim3(8, 4, BATCH), 32, 0, stream>>>(
      Qb, Kbf, Vtb, mask, ctx, 0, NVV, 0, NVV, 0, 4, 128, 7, 0);
  // r2c: heads 4-7, context self-attn
  attn_kernel<<<dim3(4, 4, BATCH), 32, 0, stream>>>(
      Qb, Kbf, Vtb, mask, ctx, NVV, NCC, NVV, NCC, VCOLCTX, 4, 64, 6, 0);

  // ---- output projection (f32 out) ----
  gemm_bf16_kernel<0, false, true, false><<<gD, blk, 0, stream>>>(
      ctx, wob, bo, ao, nullptr, DMODEL, DMODEL);

  // ---- h = LN(x + attn_out) ----
  residual_ln_kernel<<<MROWS, 256, 0, stream>>>(x, ao, g1, be1, hF, hB);

  // ---- FFN ----
  gemm_bf16_kernel<0, true, false, true><<<gF, blk, 0, stream>>>(
      hB, w1b, b1, nullptr, f1b, DFF, DMODEL);
  gemm_bf16_kernel<0, false, true, false><<<gD, blk, 0, stream>>>(
      f1b, w2b, b2, ffF, nullptr, DMODEL, DFF);

  // ---- out = LN(h + ff) ----
  residual_ln_kernel<<<MROWS, 256, 0, stream>>>(hF, ffF, g2, be2,
                                                (float*)d_out, nullptr);
}
